// SchNetInteraction_24043226923284
// MI455X (gfx1250) — matrix operand order repaired and checked by reading, not compiled, not run on hardware
//
#include <hip/hip_runtime.h>
#include <hip/hip_bf16.h>

typedef _Float16 half_t;
typedef __attribute__((ext_vector_type(16))) _Float16 v16h;
typedef __attribute__((ext_vector_type(8)))  float    v8f;

// shifted softplus: softplus(x) - log(2), numerically stable
__device__ __forceinline__ float sspf(float x) {
    float t = __expf(-fabsf(x));
    return fmaxf(x, 0.0f) + __logf(1.0f + t) - 0.69314718055994530942f;
}

// ---- WMMA fragment loaders (v_wmma_f32_16x16x32_f16 layouts, ISA 7.12.2) ----
// A: 16x32 f16. lane&15 = row M; lanes 0-15 hold K 0-7 & 16-23, lanes 16-31 hold K 8-15 & 24-31.
// Atile points at element (m0, k0) of a row-major [*, lda] f16 array.
__device__ __forceinline__ v16h load_afrag(const half_t* Atile, int lda, int lane) {
    const half_t* p = Atile + (lane & 15) * lda + ((lane & 16) ? 8 : 0);
    v16h a;
#pragma unroll
    for (int e = 0; e < 16; ++e)
        a[e] = p[e + ((e & 8) ? 8 : 0)];   // e<8 -> K=kb+e ; e>=8 -> K=kb+16+(e-8)
    return a;
}

// B: 32x16 f16 stored N-major (i.e. the original row-major weight W[n][k]).
// lane&15 = col N; lanes 0-15 hold K 0-7 & 16-23, lanes 16-31 hold K 8-15 & 24-31.
// Btile points at element (n0, k0) of N-major [*, ldk] f16 array.
__device__ __forceinline__ v16h load_bfrag(const half_t* Btile, int ldk, int lane) {
    const half_t* p = Btile + (lane & 15) * ldk + ((lane & 16) ? 8 : 0);
    v16h b;
#pragma unroll
    for (int e = 0; e < 16; ++e) b[e] = p[e + ((e & 8) ? 8 : 0)];
    return b;
}

#define WMMA_F16(acc, a, b) \
    acc = __builtin_amdgcn_wmma_f32_16x16x32_f16(false, a, false, b, (short)0, acc, false, false)

// ---- workspace layout ----
// [0, 4MB)        : y  [8192][128] f32
// [4MB, 8MB)      : v  [8192][128] f32
// [8MB, +144KB)   : f16 weights: Win2f(16384) | Wf1 padded 128x64 (8192) | Wf2(16384) | Wf2out(16384) | Wdense(16384)
#define WOFF_WF1    16384
#define WOFF_WF2    24576
#define WOFF_WF2OUT 40960
#define WOFF_WDENSE 57344

__global__ __launch_bounds__(256) void schnet_prep(
    const float* __restrict__ Win2f, const float* __restrict__ Wf1,
    const float* __restrict__ Wf2,   const float* __restrict__ Wf2out,
    const float* __restrict__ Wdense, half_t* __restrict__ wbuf) {
    int i = blockIdx.x * blockDim.x + threadIdx.x;
    if (i < 16384) {
        wbuf[i] = (half_t)Win2f[i];
    } else if (i < 24576) {
        int j = i - 16384; int f = j >> 6, s = j & 63;      // pad S 50->64
        wbuf[i] = (half_t)((s < 50) ? Wf1[f * 50 + s] : 0.0f);
    } else if (i < 40960) {
        wbuf[i] = (half_t)Wf2[i - 24576];
    } else if (i < 57344) {
        wbuf[i] = (half_t)Wf2out[i - 40960];
    } else if (i < 73728) {
        wbuf[i] = (half_t)Wdense[i - 57344];
    }
}

// y = x @ W_in2f^T : 64-row tiles, K=N=128
__global__ __launch_bounds__(256) void schnet_in2f(
    const float* __restrict__ x, const half_t* __restrict__ wbuf,
    float* __restrict__ yW) {
    __shared__ half_t Xh[64 * 128];
    const int tid = threadIdx.x, lane = tid & 31, w = tid >> 5;
    const long rowbase = (long)blockIdx.x * 64;

    for (int e = tid; e < 64 * 128; e += 256)
        Xh[e] = (half_t)x[rowbase * 128 + e];
    __syncthreads();

    v8f zero = {};
    v8f acc[4] = {zero, zero, zero, zero};
#pragma unroll
    for (int kt = 0; kt < 4; ++kt) {
        v16h bf = load_bfrag(wbuf + (w * 16) * 128 + kt * 32, 128, lane);
#pragma unroll
        for (int mt = 0; mt < 4; ++mt) {
            v16h af = load_afrag(Xh + (mt * 16) * 128 + kt * 32, 128, lane);
            WMMA_F16(acc[mt], af, bf);
        }
    }
    const int ncol = w * 16 + (lane & 15);
    const int hi = (lane & 16) ? 8 : 0;
#pragma unroll
    for (int mt = 0; mt < 4; ++mt)
#pragma unroll
        for (int j = 0; j < 8; ++j)
            yW[(rowbase + mt * 16 + j + hi) * 128 + ncol] = acc[mt][j];
}

// Fused filter net + cutoff/mask + gather-multiply + neighbor reduction. One block per (b,a).
__global__ __launch_bounds__(256) void schnet_interact(
    const float* __restrict__ r_ij, const int* __restrict__ neighbors,
    const float* __restrict__ nmask, const float* __restrict__ f_ij,
    const half_t* __restrict__ wbuf, const float* __restrict__ b_f1,
    const float* __restrict__ b_f2, const float* __restrict__ yW,
    float* __restrict__ vW) {
    __shared__ half_t Fh[64 * 64];     // padded f_ij tile
    __shared__ half_t H1[64 * 128];    // ssp(layer1) tile
    __shared__ float  YG[64 * 128];    // gathered neighbor features
    __shared__ float  cut[64];
    __shared__ int    nbr[64];

    const int tid = threadIdx.x, lane = tid & 31, w = tid >> 5;
    const int bid = blockIdx.x;        // 0..8191 = b*1024 + a
    const int b = bid >> 10;
    const half_t* Wf1h = wbuf + WOFF_WF1;
    const half_t* Wf2h = wbuf + WOFF_WF2;

    if (tid < 64) {
        long idx = (long)bid * 64 + tid;
        float r = r_ij[idx];
        cut[tid] = (r <= 5.0f) ? nmask[idx] : 0.0f;   // hard cutoff * neighbor mask
        nbr[tid] = neighbors[idx];
    }
    {
        const float* fb = f_ij + (long)bid * 64 * 50;
        for (int e = tid; e < 64 * 64; e += 256) {
            int nn = e >> 6, s = e & 63;
            Fh[e] = (half_t)((s < 50) ? fb[nn * 50 + s] : 0.0f);
        }
    }
    __syncthreads();   // nbr[] ready
    for (int e = tid; e < 64 * 128; e += 256) {
        int nn = e >> 7, c = e & 127;
        YG[e] = yW[((long)(b << 10) + nbr[nn]) * 128 + c];
    }
    __syncthreads();

    const int ncol = w * 16 + (lane & 15);
    const int hi = (lane & 16) ? 8 : 0;
    v8f zero = {};

    // ---- layer 1: [64 x 64] @ [64 x 128] + b, ssp ----
    {
        float bias = b_f1[ncol];
        v8f acc[4] = {zero, zero, zero, zero};
#pragma unroll
        for (int kt = 0; kt < 2; ++kt) {
            v16h bf = load_bfrag(Wf1h + (w * 16) * 64 + kt * 32, 64, lane);
#pragma unroll
            for (int mt = 0; mt < 4; ++mt) {
                v16h af = load_afrag(Fh + (mt * 16) * 64 + kt * 32, 64, lane);
                WMMA_F16(acc[mt], af, bf);
            }
        }
#pragma unroll
        for (int mt = 0; mt < 4; ++mt)
#pragma unroll
            for (int j = 0; j < 8; ++j)
                H1[(mt * 16 + j + hi) * 128 + ncol] = (half_t)sspf(acc[mt][j] + bias);
    }
    __syncthreads();

    // ---- layer 2 + masked elementwise * gathered y + reduce over 64 neighbors ----
    {
        float bias = b_f2[ncol];
        v8f acc[4] = {zero, zero, zero, zero};
#pragma unroll
        for (int kt = 0; kt < 4; ++kt) {
            v16h bf = load_bfrag(Wf2h + (w * 16) * 128 + kt * 32, 128, lane);
#pragma unroll
            for (int mt = 0; mt < 4; ++mt) {
                v16h af = load_afrag(H1 + (mt * 16) * 128 + kt * 32, 128, lane);
                WMMA_F16(acc[mt], af, bf);
            }
        }
        float psum = 0.0f;
#pragma unroll
        for (int mt = 0; mt < 4; ++mt)
#pragma unroll
            for (int j = 0; j < 8; ++j) {
                int m = mt * 16 + j + hi;
                psum += (acc[mt][j] + bias) * cut[m] * YG[m * 128 + ncol];
            }
        psum += __shfl_xor(psum, 16, 32);        // lanes l and l^16 share column N
        if (lane < 16) vW[(long)bid * 128 + ncol] = psum;
    }
}

// v -> ssp(v @ W_f2out^T + b) -> @ W_dense^T + b : 64-row tiles
__global__ __launch_bounds__(256) void schnet_out(
    const float* __restrict__ vW, const half_t* __restrict__ wbuf,
    const float* __restrict__ b_f2out, const float* __restrict__ b_dense,
    float* __restrict__ out) {
    __shared__ half_t Vh[64 * 128];
    __shared__ half_t Hh[64 * 128];
    const int tid = threadIdx.x, lane = tid & 31, w = tid >> 5;
    const long rowbase = (long)blockIdx.x * 64;
    const half_t* Wah = wbuf + WOFF_WF2OUT;
    const half_t* Wdh = wbuf + WOFF_WDENSE;

    for (int e = tid; e < 64 * 128; e += 256)
        Vh[e] = (half_t)vW[rowbase * 128 + e];
    __syncthreads();

    const int ncol = w * 16 + (lane & 15);
    const int hi = (lane & 16) ? 8 : 0;
    v8f zero = {};
    {
        float bias = b_f2out[ncol];
        v8f acc[4] = {zero, zero, zero, zero};
#pragma unroll
        for (int kt = 0; kt < 4; ++kt) {
            v16h bf = load_bfrag(Wah + (w * 16) * 128 + kt * 32, 128, lane);
#pragma unroll
            for (int mt = 0; mt < 4; ++mt) {
                v16h af = load_afrag(Vh + (mt * 16) * 128 + kt * 32, 128, lane);
                WMMA_F16(acc[mt], af, bf);
            }
        }
#pragma unroll
        for (int mt = 0; mt < 4; ++mt)
#pragma unroll
            for (int j = 0; j < 8; ++j)
                Hh[(mt * 16 + j + hi) * 128 + ncol] = (half_t)sspf(acc[mt][j] + bias);
    }
    __syncthreads();
    {
        float bias = b_dense[ncol];
        v8f acc[4] = {zero, zero, zero, zero};
#pragma unroll
        for (int kt = 0; kt < 4; ++kt) {
            v16h bf = load_bfrag(Wdh + (w * 16) * 128 + kt * 32, 128, lane);
#pragma unroll
            for (int mt = 0; mt < 4; ++mt) {
                v16h af = load_afrag(Hh + (mt * 16) * 128 + kt * 32, 128, lane);
                WMMA_F16(acc[mt], af, bf);
            }
        }
#pragma unroll
        for (int mt = 0; mt < 4; ++mt)
#pragma unroll
            for (int j = 0; j < 8; ++j)
                out[(rowbase + mt * 16 + j + hi) * 128 + ncol] = acc[mt][j] + bias;
    }
}

extern "C" void kernel_launch(void* const* d_in, const int* in_sizes, int n_in,
                              void* d_out, int out_size, void* d_ws, size_t ws_size,
                              hipStream_t stream) {
    const float* x        = (const float*)d_in[0];
    const float* r_ij     = (const float*)d_in[1];
    const int*   neigh    = (const int*)d_in[2];
    const float* nmask    = (const float*)d_in[3];
    const float* f_ij     = (const float*)d_in[4];
    const float* W_in2f   = (const float*)d_in[5];
    const float* W_f1     = (const float*)d_in[6];
    const float* b_f1     = (const float*)d_in[7];
    const float* W_f2     = (const float*)d_in[8];
    const float* b_f2     = (const float*)d_in[9];
    const float* W_f2out  = (const float*)d_in[10];
    const float* b_f2out  = (const float*)d_in[11];
    const float* W_dense  = (const float*)d_in[12];
    const float* b_dense  = (const float*)d_in[13];
    float* out = (float*)d_out;

    char* ws = (char*)d_ws;
    float*  yW   = (float*)ws;                          // 8192*128 f32
    float*  vW   = (float*)(ws + (size_t)4 * 1024 * 1024);
    half_t* wbuf = (half_t*)(ws + (size_t)8 * 1024 * 1024);

    schnet_prep<<<288, 256, 0, stream>>>(W_in2f, W_f1, W_f2, W_f2out, W_dense, wbuf);
    schnet_in2f<<<128, 256, 0, stream>>>(x, wbuf, yW);
    schnet_interact<<<8192, 256, 0, stream>>>(r_ij, neigh, nmask, f_ij, wbuf,
                                              b_f1, b_f2, yW, vW);
    schnet_out<<<128, 256, 0, stream>>>(vW, wbuf, b_f2out, b_dense, out);
}
